// imagetoclass2_71219147702668
// MI455X (gfx1250) — matrix operand order repaired and checked by reading, not compiled, not run on hardware
//
#include <hip/hip_runtime.h>
#include <hip/hip_bf16.h>
#include <math.h>

// ---- CDNA5 WMMA types -------------------------------------------------------
typedef __attribute__((ext_vector_type(16))) _Float16 v16h;
typedef __attribute__((ext_vector_type(8)))  _Float16 v8h;
typedef __attribute__((ext_vector_type(8)))  float    v8f;

#define N_CLASS 16
#define SUP_S   8      // samples per class (128/16)
#define C_DIM   64     // channels (K)
#define PSPAT   25     // 5*5 spatial positions
#define M_DIM   200    // SUP_S * PSPAT descriptors per class
#define M_PAD   208    // 13 tiles of 16
#define P_PAD   32     // 2 tiles of 16
#define NQ      1024
#define QB      8      // queries per workgroup (1 per wave)
#define KTOP    3
#define NEG_INF (-3.0e38f)

// Branchless sorted top-3 insert (t0 >= t1 >= t2, descending like top_k).
// 5 min/max VALU ops, no EXEC-mask branching.
__device__ __forceinline__ void top3_insert(float v, float& t0, float& t1, float& t2) {
    float a = fmaxf(t0, v);
    float b = fminf(t0, v);
    float c = fmaxf(t1, b);
    float d = fminf(t1, b);
    float e = fmaxf(t2, d);
    t0 = a; t1 = c; t2 = e;
}

__global__ __launch_bounds__(256)
void imagetoclass_wmma_kernel(const float* __restrict__ support,
                              const float* __restrict__ query,
                              float* __restrict__ out) {
    // LDS staging: normalized f16 operands, laid out for b128 fragment loads.
    // suplds[m][k]  : row-major, k contiguous (stride 64 f16 = 128 B)
    // qlds[qi][p][k]: row-major, k contiguous
    __shared__ _Float16 suplds[M_PAD * C_DIM];        // 26,624 B
    __shared__ _Float16 qlds[QB * P_PAD * C_DIM];     // 32,768 B

    const int tid    = threadIdx.x;
    const int lane   = tid & 31;
    const int wave   = tid >> 5;           // 0..7 -> query within block
    const int cls    = blockIdx.y;         // 0..15
    const int qblock = blockIdx.x;         // 0..127
    const int q      = qblock * QB + wave;

    // ---- Stage + L2-normalize support class `cls` into LDS (f16) ----------
    // support element (sample, c, p) at ((cls*8 + sample)*64 + c)*25 + p
    if (tid < M_PAD) {
        const int m = tid;
        if (m < M_DIM) {
            const int sample = m / PSPAT;
            const int hw     = m % PSPAT;
            const size_t base = ((size_t)(cls * SUP_S + sample) * C_DIM) * PSPAT + hw;
            float vals[C_DIM];
            float ss = 0.0f;
            #pragma unroll 8
            for (int c = 0; c < C_DIM; ++c) {
                float v = support[base + (size_t)c * PSPAT];
                vals[c] = v;
                ss += v * v;
            }
            const float rn = rsqrtf(ss);
            #pragma unroll 8
            for (int c = 0; c < C_DIM; ++c)
                suplds[m * C_DIM + c] = (_Float16)(vals[c] * rn);
        } else {
            for (int c = 0; c < C_DIM; ++c)
                suplds[m * C_DIM + c] = (_Float16)0.0f;
        }
    }

    // ---- Stage + L2-normalize 8 queries into LDS (f16) ---------------------
    // thread t = qi*32 + p  (256 threads exactly)
    {
        const int qi = tid >> 5;
        const int p  = tid & 31;
        const int qq = qblock * QB + qi;
        _Float16* dst = &qlds[(qi * P_PAD + p) * C_DIM];
        if (p < PSPAT) {
            const size_t base = ((size_t)qq * C_DIM) * PSPAT + p;
            float vals[C_DIM];
            float ss = 0.0f;
            #pragma unroll 8
            for (int c = 0; c < C_DIM; ++c) {
                float v = query[base + (size_t)c * PSPAT];
                vals[c] = v;
                ss += v * v;
            }
            const float rn = rsqrtf(ss);
            #pragma unroll 8
            for (int c = 0; c < C_DIM; ++c)
                dst[c] = (_Float16)(vals[c] * rn);
        } else {
            for (int c = 0; c < C_DIM; ++c)
                dst[c] = (_Float16)0.0f;
        }
    }

    __syncthreads();

    // ---- Fragment addressing (per ISA 16-bit WMMA layouts, wave32) --------
    // A (16x32, MxK): lane L -> row M=L%16; half h=L/16:
    //   VGPR0..3 = K (8h .. 8h+7), VGPR4..7 = K (16+8h .. 16+8h+7)
    // B (32x16, KxN): lane L -> col N=L%16; half h=L/16:
    //   VGPR0..3 = K (8h .. 8h+7), VGPR4..7 = K (16+8h .. 16+8h+7)
    // C/D (16x16 f32): lane L -> col N=L%16; VGPR i -> row M = i + 8h
    const int half = lane >> 4;
    const int l16  = lane & 15;

    const _Float16* qbase = &qlds[wave * P_PAD * C_DIM];

    // B fragments for both N-tiles (P columns) and both k-steps; invariant
    // over the M loop. 8x ds_load_b128 total per wave.
    v16h bfrag[2][2];   // [ntile][kstep]
    #pragma unroll
    for (int nt = 0; nt < 2; ++nt) {
        const int pcol = nt * 16 + l16;
        const _Float16* bptr = qbase + pcol * C_DIM + 8 * half;
        v8h b00 = *(const v8h*)(bptr);          // kstep0: K 8h..8h+7
        v8h b01 = *(const v8h*)(bptr + 16);     // kstep0: K 16+8h..16+8h+7
        v8h b10 = *(const v8h*)(bptr + 32);     // kstep1
        v8h b11 = *(const v8h*)(bptr + 48);
        #pragma unroll
        for (int i = 0; i < 8; ++i) {
            bfrag[nt][0][i] = b00[i]; bfrag[nt][0][i + 8] = b01[i];
            bfrag[nt][1][i] = b10[i]; bfrag[nt][1][i + 8] = b11[i];
        }
    }

    // Running top-3 per lane, per N-tile.
    float t0[2] = {NEG_INF, NEG_INF};
    float t1[2] = {NEG_INF, NEG_INF};
    float t2[2] = {NEG_INF, NEG_INF};

    #pragma unroll
    for (int mt = 0; mt < 13; ++mt) {
        const int mrow = mt * 16 + l16;
        const _Float16* aptr = &suplds[mrow * C_DIM + 8 * half];
        v8h a00 = *(const v8h*)(aptr);          // kstep0: K 8h..8h+7
        v8h a01 = *(const v8h*)(aptr + 16);     // kstep0: K 16+8h..
        v8h a10 = *(const v8h*)(aptr + 32);     // kstep1
        v8h a11 = *(const v8h*)(aptr + 48);
        v16h afrag0, afrag1;
        #pragma unroll
        for (int i = 0; i < 8; ++i) {
            afrag0[i] = a00[i]; afrag0[i + 8] = a01[i];
            afrag1[i] = a10[i]; afrag1[i + 8] = a11[i];
        }

        // Rows this lane sees: mt*16 + i + 8*half, i=0..7. Only mtile 12 /
        // half==1 lanes hit padded rows (200..207): mask them branchlessly.
        const bool rows_valid = (mt < 12) | (half == 0);

        #pragma unroll
        for (int nt = 0; nt < 2; ++nt) {
            v8f acc = {};
            acc = __builtin_amdgcn_wmma_f32_16x16x32_f16(
                false, afrag0, false, bfrag[nt][0], (short)0, acc, false, false);
            acc = __builtin_amdgcn_wmma_f32_16x16x32_f16(
                false, afrag1, false, bfrag[nt][1], (short)0, acc, false, false);

            #pragma unroll
            for (int i = 0; i < 8; ++i) {
                const float vv = rows_valid ? acc[i] : NEG_INF;  // v_cndmask
                top3_insert(vv, t0[nt], t1[nt], t2[nt]);
            }
        }
    }

    // Merge with partner lane (same column, other row-half): wave32 xor-shuffle
    // then 3 more branchless inserts per N-tile. Lanes 0..15 hold the result.
    #pragma unroll
    for (int nt = 0; nt < 2; ++nt) {
        float p0 = __shfl_xor(t0[nt], 16);
        float p1 = __shfl_xor(t1[nt], 16);
        float p2 = __shfl_xor(t2[nt], 16);
        top3_insert(p0, t0[nt], t1[nt], t2[nt]);
        top3_insert(p1, t0[nt], t1[nt], t2[nt]);
        top3_insert(p2, t0[nt], t1[nt], t2[nt]);

        const int pcol = nt * 16 + l16;
        if (half == 0 && pcol < PSPAT) {
            const size_t row = (size_t)(cls * NQ + q);
            const size_t ofs = row * (KTOP * PSPAT) + pcol;
            out[ofs]             = t0[nt];
            out[ofs + PSPAT]     = t1[nt];
            out[ofs + 2 * PSPAT] = t2[nt];
        }
    }
}

extern "C" void kernel_launch(void* const* d_in, const int* in_sizes, int n_in,
                              void* d_out, int out_size, void* d_ws, size_t ws_size,
                              hipStream_t stream) {
    (void)in_sizes; (void)n_in; (void)d_ws; (void)ws_size; (void)out_size;
    const float* support = (const float*)d_in[0];
    const float* query   = (const float*)d_in[1];
    float* out = (float*)d_out;

    dim3 grid(NQ / QB, N_CLASS);   // 128 x 16 workgroups
    dim3 block(256);               // 8 waves (wave32): 1 query per wave
    imagetoclass_wmma_kernel<<<grid, block, 0, stream>>>(support, query, out);
}
